// ConditionalGarmentDrapingModel_81956565943018
// MI455X (gfx1250) — compile-verified, matrix operand order checked by reading, not run on hardware
//
#include <hip/hip_runtime.h>

#define NODE_H 128
#define COND_C 98
#define NLAYER 4
#define LN_EPS 1e-5f

typedef __attribute__((ext_vector_type(16))) __bf16 v16bf;
typedef __attribute__((ext_vector_type(8)))  float  v8f;

static __device__ __forceinline__ unsigned short f2bf(float f) {
  union { float f; unsigned int u; } v; v.f = f;
  unsigned int r = v.u + 0x7FFFu + ((v.u >> 16) & 1u);
  return (unsigned short)(r >> 16);
}

// ---------------------------------------------------------------------------
// CSR build
// ---------------------------------------------------------------------------
__global__ void zero_i32_kernel(int* __restrict__ p, int n) {
  int i = blockIdx.x * blockDim.x + threadIdx.x;
  if (i < n) p[i] = 0;
}

__global__ void degree_kernel(const long long* __restrict__ ei, int* __restrict__ deg,
                              int E, int N) {
  int e = blockIdx.x * blockDim.x + threadIdx.x;
  if (e < E) {
    int dst = (int)ei[(size_t)E + e];
    if (dst >= 0 && dst < N) atomicAdd(&deg[dst], 1);
  }
}

__global__ __launch_bounds__(1024)
void scan_kernel(const int* __restrict__ deg, int* __restrict__ row_ptr,
                 float* __restrict__ count_f, int N) {
  __shared__ int sc[1024];
  __shared__ int carry;
  const int t = threadIdx.x;
  if (t == 0) { carry = 0; row_ptr[0] = 0; }
  __syncthreads();
  for (int base = 0; base < N; base += 1024) {
    int i = base + t;
    int v = (i < N) ? deg[i] : 0;
    if (i < N) count_f[i] = (float)((v < 1) ? 1 : v);
    sc[t] = v;
    __syncthreads();
    for (int off = 1; off < 1024; off <<= 1) {
      int add = (t >= off) ? sc[t - off] : 0;
      __syncthreads();
      sc[t] += add;
      __syncthreads();
    }
    int incl = sc[t];
    if (i < N) row_ptr[i + 1] = carry + incl;
    __syncthreads();
    if (t == 1023) carry += incl;
    __syncthreads();
  }
}

__global__ void fill_kernel(const long long* __restrict__ ei, const int* __restrict__ row_ptr,
                            int* __restrict__ fill, int* __restrict__ csr, int E, int N) {
  int e = blockIdx.x * blockDim.x + threadIdx.x;
  if (e < E) {
    int src = (int)ei[e];
    int dst = (int)ei[(size_t)E + e];
    if (dst >= 0 && dst < N) {
      int p = atomicAdd(&fill[dst], 1);
      csr[row_ptr[dst] + p] = src;
    }
  }
}

// ---------------------------------------------------------------------------
// FiLM prep: gamma = cond@Wg + bg ; beta2 = gamma*bc + (cond@Wb + bb)
// ---------------------------------------------------------------------------
__global__ __launch_bounds__(128)
void film_kernel(const float* __restrict__ smpl, const float* __restrict__ tmpl,
                 const float* __restrict__ Wg, const float* __restrict__ bg,
                 const float* __restrict__ Wb, const float* __restrict__ bb,
                 const float* __restrict__ bc,
                 float* __restrict__ gamma, float* __restrict__ beta2) {
  __shared__ float cond[COND_C];
  const int t = threadIdx.x;
  if (t < 82) cond[t] = smpl[t];
  else if (t < COND_C) cond[t] = tmpl[t - 82];
  __syncthreads();
  const int c = t;  // 0..127
  for (int l = 0; l < NLAYER; ++l) {
    float g  = bg[l * NODE_H + c];
    float bt = bb[l * NODE_H + c];
    for (int j = 0; j < COND_C; ++j) {
      float cj = cond[j];
      g  += cj * Wg[((size_t)l * COND_C + j) * NODE_H + c];
      bt += cj * Wb[((size_t)l * COND_C + j) * NODE_H + c];
    }
    gamma[l * NODE_H + c] = g;
    beta2[l * NODE_H + c] = g * bc[l * NODE_H + c] + bt;
  }
}

// ---------------------------------------------------------------------------
// Weight swizzle: f32 -> bf16 in WMMA B-fragment order
// layout [kb][nt][lane][16 ushort]; within frag: idx = 8*((k>>4)&1) + (k&7),
// lane = n%16 + 16*((k>>3)&1)
// ---------------------------------------------------------------------------
__global__ void swz_kernel(const float* __restrict__ Wc, const float* __restrict__ Wh1,
                           unsigned short* __restrict__ WcSw, unsigned short* __restrict__ W1Sw) {
  int i = blockIdx.x * blockDim.x + threadIdx.x;
  const int totC = NLAYER * 16384;
  if (i < totC) {
    int l = i >> 14, r = i & 16383;
    int vec = r >> 4, idx = r & 15;
    int lane = vec & 31, nt = (vec >> 5) & 7, kb = vec >> 8;
    int k = (kb << 5) + ((idx >> 3) << 4) + ((lane >> 4) << 3) + (idx & 7);
    int n = (nt << 4) + (lane & 15);
    WcSw[i] = f2bf(Wc[((size_t)l << 14) + (size_t)k * NODE_H + n]);
  } else if (i < totC + 8192) {
    int j = i - totC;
    int vec = j >> 4, idx = j & 15;
    int lane = vec & 31, nt = (vec >> 5) & 3, kb = vec >> 7;
    int k = (kb << 5) + ((idx >> 3) << 4) + ((lane >> 4) << 3) + (idx & 7);
    int n = (nt << 4) + (lane & 15);
    W1Sw[j] = f2bf(Wh1[(size_t)k * 64 + n]);
  }
}

// ---------------------------------------------------------------------------
// Input embedding: h0 = relu(nf @ Wi + bi)
// ---------------------------------------------------------------------------
__global__ __launch_bounds__(256)
void input_kernel(const float* __restrict__ nf, const float* __restrict__ Wi,
                  const float* __restrict__ bi, float* __restrict__ h, int N) {
  __shared__ float w[3][NODE_H];
  __shared__ float bsh[NODE_H];
  const int t = threadIdx.x;
  for (int i = t; i < 512; i += 256) {
    if (i < 384) w[i / NODE_H][i & 127] = Wi[i];
    else bsh[i - 384] = bi[i - 384];
  }
  __syncthreads();
  size_t idx = (size_t)blockIdx.x * 256 + t;
  if (idx < (size_t)N * NODE_H) {
    int n = (int)(idx >> 7), c = (int)(idx & 127);
    float v = bsh[c] + nf[n * 3 + 0] * w[0][c] + nf[n * 3 + 1] * w[1][c] + nf[n * 3 + 2] * w[2][c];
    h[idx] = (v > 0.f) ? v : 0.f;
  }
}

// ---------------------------------------------------------------------------
// Fused GNN layer: mean-aggregate -> WMMA GEMM -> FiLM -> relu -> residual -> LN
// One block = 16 node rows, 128 threads (4 waves of 32).
// ---------------------------------------------------------------------------
__global__ __launch_bounds__(128)
void layer_kernel(const float* __restrict__ hin, float* __restrict__ hout,
                  const int* __restrict__ row_ptr, const int* __restrict__ csr,
                  const float* __restrict__ count_f,
                  const v16bf* __restrict__ Wsw,       // [kb4][nt8][lane32]
                  const float* __restrict__ gamma, const float* __restrict__ beta2,
                  const float* __restrict__ lng, const float* __restrict__ lnb,
                  int N) {
  __shared__ __align__(32) unsigned short As[4][32][16];  // bf16 A fragments
  __shared__ float hs[16][NODE_H];   // f32 tile of h (for residual)
  __shared__ float ts[16][NODE_H];   // pre-LN values
  __shared__ float redS[16][8];
  __shared__ float redQ[16][8];

  const int t = threadIdx.x;
  const int lane = t & 31;
  const int wave = t >> 5;
  const int base = blockIdx.x * 16;

  // stage tile rows of h
  for (int i = t; i < 16 * NODE_H; i += 128) {
    int r = i >> 7, c = i & 127;
    int node = base + r;
    hs[r][c] = (node < N) ? hin[(size_t)node * NODE_H + c] : 0.0f;
  }
  __syncthreads();

  // mean aggregation: each wave handles 4 rows, each lane 4 contiguous columns
  for (int rr = 0; rr < 4; ++rr) {
    int r = wave * 4 + rr;
    int node = base + r;
    float a0 = 0.f, a1 = 0.f, a2 = 0.f, a3 = 0.f;
    if (node < N) {
      int s = row_ptr[node], e = row_ptr[node + 1];
      for (int j = s; j < e; ++j) {
        int srcn = csr[j];
        if (j + 1 < e)
          __builtin_prefetch(&hin[(size_t)csr[j + 1] * NODE_H + lane * 4], 0, 1);
        const float4 hv = *(const float4*)(hin + (size_t)srcn * NODE_H + lane * 4);
        a0 += hv.x; a1 += hv.y; a2 += hv.z; a3 += hv.w;
      }
      float inv = 1.0f / count_f[node];
      a0 *= inv; a1 *= inv; a2 *= inv; a3 *= inv;
    }
    const int c0 = lane * 4;
    float av[4] = {a0, a1, a2, a3};
#pragma unroll
    for (int q = 0; q < 4; ++q) {
      int c = c0 + q;
      float x = hs[r][c] + av[q];   // h + agg/count
      As[c >> 5][r + (((c >> 3) & 1) << 4)][(((c >> 4) & 1) << 3) + (c & 7)] = f2bf(x);
    }
  }
  __syncthreads();

  // WMMA: (h+agg)[16x128] @ Wc[128x128], each wave owns 2 column tiles
#pragma unroll
  for (int i = 0; i < 2; ++i) {
    const int nt = wave * 2 + i;
    v8f acc = {};
#pragma unroll
    for (int kb = 0; kb < 4; ++kb) {
      v16bf a = *(const v16bf*)(&As[kb][lane][0]);
      v16bf b = Wsw[((kb << 3) + nt) * 32 + lane];
      acc = __builtin_amdgcn_wmma_f32_16x16x32_bf16(false, a, false, b,
                                                    (short)0, acc, false, false);
    }
    const int ncol = (nt << 4) + (lane & 15);
    const float g  = gamma[ncol];
    const float bt = beta2[ncol];   // gamma*bc + beta folded
    const int mb = (lane >> 4) << 3;
#pragma unroll
    for (int r = 0; r < 8; ++r) {
      int m = mb + r;
      float v = g * acc[r] + bt;    // FiLM
      v = (v > 0.f) ? v : 0.f;      // relu
      ts[m][ncol] = hs[m][ncol] + v;  // residual
    }
  }
  __syncthreads();

  // row-wise LayerNorm: 8 threads per row of 128
  {
    const int row = t >> 3, seg = t & 7;
    float s = 0.f, q = 0.f;
#pragma unroll
    for (int j = 0; j < 16; ++j) {
      float v = ts[row][(seg << 4) + j];
      s += v; q += v * v;
    }
    redS[row][seg] = s; redQ[row][seg] = q;
    __syncthreads();
    float su = 0.f, sq = 0.f;
#pragma unroll
    for (int j = 0; j < 8; ++j) { su += redS[row][j]; sq += redQ[row][j]; }
    const float mu  = su * (1.0f / NODE_H);
    const float var = sq * (1.0f / NODE_H) - mu * mu;
    const float inv = rsqrtf(var + LN_EPS);
    const int node = base + row;
    if (node < N) {
#pragma unroll
      for (int j = 0; j < 16; ++j) {
        int c = (seg << 4) + j;
        hout[(size_t)node * NODE_H + c] = (ts[row][c] - mu) * inv * lng[c] + lnb[c];
      }
    }
  }
}

// ---------------------------------------------------------------------------
// Head: out = relu(h @ Wh1 + bh1) @ Wh2 + bh2
// ---------------------------------------------------------------------------
__global__ __launch_bounds__(128)
void head_kernel(const float* __restrict__ hin,
                 const v16bf* __restrict__ W1sw,   // [kb4][nt4][lane32]
                 const float* __restrict__ bh1,
                 const float* __restrict__ Wh2, const float* __restrict__ bh2,
                 float* __restrict__ out, int N) {
  __shared__ __align__(32) unsigned short As[4][32][16];
  __shared__ float rs[16][64];
  const int t = threadIdx.x, lane = t & 31, wave = t >> 5;
  const int base = blockIdx.x * 16;

  for (int i = t; i < 16 * NODE_H; i += 128) {
    int r = i >> 7, c = i & 127;
    int node = base + r;
    float x = (node < N) ? hin[(size_t)node * NODE_H + c] : 0.f;
    As[c >> 5][r + (((c >> 3) & 1) << 4)][(((c >> 4) & 1) << 3) + (c & 7)] = f2bf(x);
  }
  __syncthreads();

  {
    const int nt = wave;  // 4 waves -> 4 column tiles of 16 (64 outputs)
    v8f acc = {};
#pragma unroll
    for (int kb = 0; kb < 4; ++kb) {
      v16bf a = *(const v16bf*)(&As[kb][lane][0]);
      v16bf b = W1sw[((kb << 2) + nt) * 32 + lane];
      acc = __builtin_amdgcn_wmma_f32_16x16x32_bf16(false, a, false, b,
                                                    (short)0, acc, false, false);
    }
    const int ncol = (nt << 4) + (lane & 15);
    const float b1 = bh1[ncol];
    const int mb = (lane >> 4) << 3;
#pragma unroll
    for (int r = 0; r < 8; ++r) {
      float v = acc[r] + b1;
      rs[mb + r][ncol] = (v > 0.f) ? v : 0.f;
    }
  }
  __syncthreads();

  if (t < 48) {
    const int m = t / 3, o = t % 3;
    float s = bh2[o];
    for (int j = 0; j < 64; ++j) s += rs[m][j] * Wh2[j * 3 + o];
    const int node = base + m;
    if (node < N) out[(size_t)node * 3 + o] = s;
  }
}

// ---------------------------------------------------------------------------
extern "C" void kernel_launch(void* const* d_in, const int* in_sizes, int n_in,
                              void* d_out, int out_size, void* d_ws, size_t ws_size,
                              hipStream_t stream) {
  const float*     nf   = (const float*)d_in[0];
  const long long* ei   = (const long long*)d_in[1];
  const float*     smpl = (const float*)d_in[2];
  const float*     tmpl = (const float*)d_in[3];
  // d_in[4] = batch_index (unused)
  const float* Wi  = (const float*)d_in[5];
  const float* bi  = (const float*)d_in[6];
  const float* Wc  = (const float*)d_in[7];
  const float* bc  = (const float*)d_in[8];
  const float* Wg  = (const float*)d_in[9];
  const float* bg  = (const float*)d_in[10];
  const float* Wb  = (const float*)d_in[11];
  const float* bb  = (const float*)d_in[12];
  const float* lng = (const float*)d_in[13];
  const float* lnb = (const float*)d_in[14];
  const float* Wh1 = (const float*)d_in[15];
  const float* bh1 = (const float*)d_in[16];
  const float* Wh2 = (const float*)d_in[17];
  const float* bh2 = (const float*)d_in[18];
  (void)n_in; (void)out_size; (void)ws_size;

  const int N = in_sizes[0] / 3;
  const int E = in_sizes[1] / 2;

  char* wsb = (char*)d_ws;
  size_t off = 0;
  auto take = [&](size_t bytes) -> void* {
    void* p = wsb + off;
    off = (off + bytes + 255) & ~(size_t)255;
    return p;
  };
  float* hA      = (float*)take((size_t)N * NODE_H * 4);
  float* hB      = (float*)take((size_t)N * NODE_H * 4);
  int*   deg     = (int*)take((size_t)N * 2 * 4);   // deg + fill, zeroed together
  int*   fill    = deg + N;
  int*   row_ptr = (int*)take(((size_t)N + 1) * 4);
  float* cntf    = (float*)take((size_t)N * 4);
  int*   csr     = (int*)take((size_t)E * 4);
  float* gamma   = (float*)take((size_t)NLAYER * NODE_H * 4);
  float* beta2   = (float*)take((size_t)NLAYER * NODE_H * 4);
  unsigned short* WcSw = (unsigned short*)take((size_t)NLAYER * 16384 * 2);
  unsigned short* W1Sw = (unsigned short*)take((size_t)8192 * 2);

  // CSR build (per-call, deterministic)
  zero_i32_kernel<<<(2 * N + 255) / 256, 256, 0, stream>>>(deg, 2 * N);
  degree_kernel<<<(E + 255) / 256, 256, 0, stream>>>(ei, deg, E, N);
  scan_kernel<<<1, 1024, 0, stream>>>(deg, row_ptr, cntf, N);
  fill_kernel<<<(E + 255) / 256, 256, 0, stream>>>(ei, row_ptr, fill, csr, E, N);

  // parameter prep
  film_kernel<<<1, 128, 0, stream>>>(smpl, tmpl, Wg, bg, Wb, bb, bc, gamma, beta2);
  {
    const int total = NLAYER * 16384 + 8192;
    swz_kernel<<<(total + 255) / 256, 256, 0, stream>>>(Wc, Wh1, WcSw, W1Sw);
  }

  // forward
  input_kernel<<<(int)(((size_t)N * NODE_H + 255) / 256), 256, 0, stream>>>(nf, Wi, bi, hA, N);

  const int nTiles = (N + 15) / 16;
  const float* cur = hA;
  float* nxt = hB;
  for (int l = 0; l < NLAYER; ++l) {
    layer_kernel<<<nTiles, 128, 0, stream>>>(
        cur, nxt, row_ptr, csr, cntf,
        (const v16bf*)(WcSw + (size_t)l * 16384),
        gamma + l * NODE_H, beta2 + l * NODE_H,
        lng + l * NODE_H, lnb + l * NODE_H, N);
    float* tmp = (float*)cur; cur = nxt; nxt = tmp;
  }
  head_kernel<<<nTiles, 128, 0, stream>>>(cur, (const v16bf*)W1Sw, bh1, Wh2, bh2,
                                          (float*)d_out, N);
}
